// RelativeAttention_33681133535771
// MI455X (gfx1250) — compile-verified, matrix operand order checked by reading, not run on hardware
//
#include <hip/hip_runtime.h>

// ---------------------------------------------------------------------------
// RelativeAttention on gfx1250 (MI455X): bf16 WMMA + pipelined async-to-LDS
//   K0: prep     x->bf16, Wqkv/Wout -> bf16 transposed [N][K]
//   K1: QKV GEMM (double-buffered async staging) -> Q,K [B,H,S,64], V^T [B,H,64,S]
//   K2: Flash attention w/ rel-pos bias (double-buffered async staging)
//   K3: Out GEMM -> d_out fp32
// ---------------------------------------------------------------------------

#define D_MODEL 1024
#define N_HEADS 16
#define HEAD_DIM 64
#define SEQ 2048
#define BATCH 2
#define MAX_LEN 2048

typedef __attribute__((ext_vector_type(16))) __bf16 v16bf;
typedef __attribute__((ext_vector_type(8)))  float  v8f;

union FragAB {
    v16bf v;
    unsigned int u[8];
};

static __device__ __forceinline__ unsigned short f2bf(float f) {
    union { float f; unsigned int u; } c; c.f = f;
    unsigned int u = c.u;
    unsigned int r = u + 0x7FFFu + ((u >> 16) & 1u);   // round-to-nearest-even
    return (unsigned short)(r >> 16);
}

static __device__ __forceinline__ v8f wmma_bf16(const FragAB& a, const FragAB& b, v8f c) {
    return __builtin_amdgcn_wmma_f32_16x16x32_bf16(
        false, a.v, false, b.v, (short)0, c, false, false);
}

// Dword index (within a 16-dword group == 32 bf16 of K) for fragment register j.
// 16-bit A/B layout (ISA 7.12.2): lanes 0-15 -> K=0..7,16..23 ; lanes 16-31 -> K=8..15,24..31
static __device__ __forceinline__ int frag_di(int j, int klo4) {
    return klo4 + (j < 4 ? j : j + 4);
}

// 16-byte async copy global->LDS (GVS mode: saddr + 32-bit voffset). ASYNCcnt-tracked.
static __device__ __forceinline__ void async_copy16(unsigned lds_addr,
                                                    const void* gbase, unsigned goff) {
    asm volatile("global_load_async_to_lds_b128 %0, %1, %2"
                 :: "v"(lds_addr), "v"(goff), "s"(gbase)
                 : "memory");
}
static __device__ __forceinline__ void wait_async0() {
    asm volatile("s_wait_asynccnt 0" ::: "memory");
}
// Wait until at most N async ops remain (async loads complete in issue order,
// so with N newly-issued copies outstanding this proves the previous tile landed).
template <int N>
static __device__ __forceinline__ void wait_asyncN() {
    asm volatile("s_wait_asynccnt %0" :: "n"(N) : "memory");
}
static __device__ __forceinline__ unsigned lds_addr_of(const void* p) {
    return (unsigned)(size_t)p;   // generic LDS address: low 32 bits = LDS offset
}

// ---------------------------------------------------------------------------
// K0: convert x to bf16 (row-major) and weights to bf16 transposed [N][K].
// ---------------------------------------------------------------------------
__global__ __launch_bounds__(256) void prep_kernel(
    const float* __restrict__ x, const float* __restrict__ Wqkv,
    const float* __restrict__ Wout,
    unsigned short* __restrict__ xb, unsigned short* __restrict__ WqkvT,
    unsigned short* __restrict__ WoutT)
{
    const int id = blockIdx.x * 256 + threadIdx.x;          // < 4M
    xb[id] = f2bf(x[id]);
    if (id < 3 * 1024 * 1024) {                             // Wqkv [1024][3072]
        const int k = id / 3072, n = id % 3072;
        WqkvT[(size_t)n * 1024 + k] = f2bf(Wqkv[id]);
    }
    if (id < 1024 * 1024) {                                 // Wout [1024][1024]
        const int k = id >> 10, n = id & 1023;
        WoutT[(size_t)n * 1024 + k] = f2bf(Wout[id]);
    }
}

// ---------------------------------------------------------------------------
// WMMA GEMM: C[M,N] = A[M,K] * BT[N,K]^T (+bias). A, BT bf16 row-major.
//  MODE 0: scatter bf16 into Q,K [B,H,S,64] and V^T [B,H,64,S]
//  MODE 1: fp32 + bias into Co
// Block 256 thr (8 waves, 4x2); tile 128x128; wave tile 32x64; K step 64;
// double-buffered async-to-LDS staging (8 copies/thread/tile).
// ---------------------------------------------------------------------------
template <int MODE>
__global__ __launch_bounds__(256) void gemm_bf16_kernel(
    const unsigned short* __restrict__ A, const unsigned short* __restrict__ BT,
    const float* __restrict__ bias,
    unsigned short* __restrict__ Qo, unsigned short* __restrict__ Ko,
    unsigned short* __restrict__ VTo, float* __restrict__ Co,
    int M, int N, int K, int nTilesN)
{
    __shared__ unsigned int Asmem[2][128 * 32];  // [buf][row][32 dwords = 64 bf16 of K]
    __shared__ unsigned int Bsmem[2][128 * 32];  // [buf][col][32 dwords = 64 bf16 of K]

    const int bm = blockIdx.x / nTilesN;
    const int bn = blockIdx.x % nTilesN;
    const int t = threadIdx.x;
    const int lane = t & 31;
    const int w = t >> 5;
    const int wm = (w & 3) * 32;
    const int wn = (w >> 2) * 64;
    const int lm = lane & 15;
    const int klo4 = (lane < 16) ? 0 : 4;
    const int hi8 = (lane < 16) ? 0 : 8;

    v8f acc[2][4];
#pragma unroll
    for (int i = 0; i < 2; ++i)
#pragma unroll
        for (int j = 0; j < 4; ++j)
            acc[i][j] = (v8f){0.f, 0.f, 0.f, 0.f, 0.f, 0.f, 0.f, 0.f};

    // stage tile ki into buffer buf: A/B rows are 128B = 8 x 16B segments
    auto stage = [&](int ki, int buf) {
        const unsigned ldsA = lds_addr_of(&Asmem[buf][0]);
        const unsigned ldsB = lds_addr_of(&Bsmem[buf][0]);
        const int k0 = ki * 64;
#pragma unroll
        for (int j = 0; j < 4; ++j) {
            const int i = t + 256 * j;
            const int row = i >> 3, seg = i & 7;
            async_copy16(ldsA + (unsigned)(row * 128 + seg * 16), A,
                         (unsigned)(((bm * 128 + row) * K + k0) * 2 + seg * 16));
        }
#pragma unroll
        for (int j = 0; j < 4; ++j) {
            const int i = t + 256 * j;
            const int col = i >> 3, seg = i & 7;
            async_copy16(ldsB + (unsigned)(col * 128 + seg * 16), BT,
                         (unsigned)(((bn * 128 + col) * K + k0) * 2 + seg * 16));
        }
    };

    const int nk = K >> 6;                      // K/64
    stage(0, 0);
    for (int ki = 0; ki < nk; ++ki) {
        const int buf = ki & 1;
        if (ki + 1 < nk) {
            stage(ki + 1, buf ^ 1);             // prefetch next tile behind compute
            wait_asyncN<8>();                   // current tile's 8 copies complete
        } else {
            wait_async0();
        }
        __syncthreads();

        const unsigned int* As = &Asmem[buf][0];
        const unsigned int* Bs = &Bsmem[buf][0];
#pragma unroll
        for (int kc = 0; kc < 2; ++kc) {        // two 32-deep K chunks
            FragAB af[2], bf[4];
#pragma unroll
            for (int ms = 0; ms < 2; ++ms) {
                const int row = wm + ms * 16 + lm;
#pragma unroll
                for (int j = 0; j < 8; ++j)
                    af[ms].u[j] = As[row * 32 + kc * 16 + frag_di(j, klo4)];
            }
#pragma unroll
            for (int ns = 0; ns < 4; ++ns) {
                const int col = wn + ns * 16 + lm;
#pragma unroll
                for (int j = 0; j < 8; ++j)
                    bf[ns].u[j] = Bs[col * 32 + kc * 16 + frag_di(j, klo4)];
            }
#pragma unroll
            for (int ms = 0; ms < 2; ++ms)
#pragma unroll
                for (int ns = 0; ns < 4; ++ns)
                    acc[ms][ns] = wmma_bf16(af[ms], bf[ns], acc[ms][ns]);
        }
        __syncthreads();                        // all reads done before buf reuse
    }

    // ---- epilogue ----
#pragma unroll
    for (int ms = 0; ms < 2; ++ms) {
#pragma unroll
        for (int ns = 0; ns < 4; ++ns) {
            const int gcol = bn * 128 + wn + ns * 16 + lm;
            const float bv = bias[gcol];
#pragma unroll
            for (int r = 0; r < 8; ++r) {
                const int grow = bm * 128 + wm + ms * 16 + r + hi8;
                const float vv = acc[ms][ns][r] + bv;
                if (MODE == 0) {
                    // gcol = which*1024 + h*64 + d ; grow = b*SEQ + s
                    const int which = gcol >> 10;
                    const int h = (gcol >> 6) & 15;
                    const int d = gcol & 63;
                    const int b = grow >> 11;
                    const int s = grow & 2047;
                    const unsigned short bvv = f2bf(vv);
                    if (which == 2) {   // V stored transposed: [b,h,d,s]
                        VTo[((size_t)(b * N_HEADS + h) * HEAD_DIM + d) * SEQ + s] = bvv;
                    } else {
                        unsigned short* dst = (which == 0) ? Qo : Ko;
                        dst[((size_t)(b * N_HEADS + h) * SEQ + s) * HEAD_DIM + d] = bvv;
                    }
                } else {
                    Co[(size_t)grow * N + gcol] = vv;
                }
            }
        }
    }
}

// ---------------------------------------------------------------------------
// K2: Flash attention with relative position bias.
// Block 128 thr (4 waves); one block per (b, h, 64-row q tile).
// Key tile = 64; K and V^T tiles double-buffered via async-to-LDS.
// ---------------------------------------------------------------------------
__global__ __launch_bounds__(128) void attn_kernel(
    const unsigned short* __restrict__ Qw, const unsigned short* __restrict__ Kw,
    const unsigned short* __restrict__ VTw, const float* __restrict__ relb,
    unsigned short* __restrict__ Ao)
{
    __shared__ unsigned int Ks32[2][64 * 32];   // [buf][key][dword d]   8KB each
    __shared__ unsigned int Vs32[2][64 * 32];   // [buf][d][dword key]   8KB each
    __shared__ unsigned int Ps[4][16 * 32];     // per-wave P: 16 x 64 bf16 (2KB each)

    const int bid = blockIdx.x;
    const int qt = bid & 31;
    const int h  = (bid >> 5) & 15;
    const int b  = bid >> 9;

    const int t = threadIdx.x;
    const int lane = t & 31;
    const int w = t >> 5;
    const int lm = lane & 15;
    const int klo4 = (lane < 16) ? 0 : 4;
    const int hi8  = (lane < 16) ? 0 : 8;

    const size_t bh = (size_t)(b * N_HEADS + h);
    const unsigned short* Qbh  = Qw  + bh * SEQ * HEAD_DIM;
    const unsigned short* Kbh  = Kw  + bh * SEQ * HEAD_DIM;
    const unsigned short* VTbh = VTw + bh * HEAD_DIM * SEQ;

    const int qbase = qt * 64 + w * 16;
    const float scale = 0.125f;  // HEAD_DIM^-0.5

    // Q fragments (16 rows x 64 d), loaded once from global.
    FragAB qf[2];
#pragma unroll
    for (int c = 0; c < 2; ++c) {
        const unsigned int* Qrow = (const unsigned int*)
            (Qbh + (size_t)(qbase + lm) * HEAD_DIM + c * 32);
#pragma unroll
        for (int j = 0; j < 8; ++j)
            qf[c].u[j] = Qrow[frag_di(j, klo4)];
    }

    // stage key-tile ti into buffer buf: 4 K copies + 4 V copies per thread
    auto stageKV = [&](int ti, int buf) {
        const int kt = ti * 64;
        const unsigned ldsK = lds_addr_of(&Ks32[buf][0]);
        const unsigned ldsV = lds_addr_of(&Vs32[buf][0]);
#pragma unroll
        for (int j = 0; j < 4; ++j) {
            const int i = t + 128 * j;
            const int key = i >> 3, seg = i & 7;    // 8 x 16B per 128B key row
            async_copy16(ldsK + (unsigned)(key * 128 + seg * 16),
                         Kbh, (unsigned)((kt + key) * 128 + seg * 16));
        }
#pragma unroll
        for (int j = 0; j < 4; ++j) {
            const int i = t + 128 * j;
            const int d = i >> 3, seg = i & 7;      // keys contiguous in V^T rows
            async_copy16(ldsV + (unsigned)(d * 128 + seg * 16),
                         VTbh, (unsigned)((d * SEQ + kt) * 2 + seg * 16));
        }
    };

    v8f o[4];
#pragma unroll
    for (int i = 0; i < 4; ++i)
        o[i] = (v8f){0.f, 0.f, 0.f, 0.f, 0.f, 0.f, 0.f, 0.f};
    float run_max[8], run_sum[8];
#pragma unroll
    for (int r = 0; r < 8; ++r) { run_max[r] = -1e30f; run_sum[r] = 0.f; }

    const int nT = SEQ / 64;                       // 32 key tiles
    stageKV(0, 0);
    for (int ti = 0; ti < nT; ++ti) {
        const int buf = ti & 1;
        const int kt = ti * 64;
        if (ti + 1 < nT) {
            stageKV(ti + 1, buf ^ 1);
            wait_asyncN<8>();
        } else {
            wait_async0();
        }
        __syncthreads();

        const unsigned int* Ksb = &Ks32[buf][0];
        const unsigned int* Vsb = &Vs32[buf][0];

        // ---- scores: S = Q K^T ; 4 key groups x 2 d-chunks = 8 WMMAs ----
        v8f sc[4];
#pragma unroll
        for (int ns = 0; ns < 4; ++ns) {
            sc[ns] = (v8f){0.f, 0.f, 0.f, 0.f, 0.f, 0.f, 0.f, 0.f};
            const int key = ns * 16 + lm;
#pragma unroll
            for (int c = 0; c < 2; ++c) {
                FragAB kf;
#pragma unroll
                for (int j = 0; j < 8; ++j)
                    kf.u[j] = Ksb[key * 32 + c * 16 + frag_di(j, klo4)];
                sc[ns] = wmma_bf16(qf[c], kf, sc[ns]);
            }
        }

        // ---- scale + rel-pos bias, per-row online softmax ----
        float corr[8];
#pragma unroll
        for (int r = 0; r < 8; ++r) {
            const int q = qbase + r + hi8;
            float m = -1e30f;
#pragma unroll
            for (int ns = 0; ns < 4; ++ns) {
                const int kk = kt + ns * 16 + lm;
                const float v = sc[ns][r] * scale + relb[q - kk + (MAX_LEN - 1)];
                sc[ns][r] = v;
                m = fmaxf(m, v);
            }
#pragma unroll
            for (int off = 1; off < 16; off <<= 1)
                m = fmaxf(m, __shfl_xor(m, off, 32));
            const float nm = fmaxf(run_max[r], m);
            corr[r] = __expf(run_max[r] - nm);
            run_max[r] = nm;
        }
#pragma unroll
        for (int r = 0; r < 8; ++r) {
            float rs = 0.f;
#pragma unroll
            for (int ns = 0; ns < 4; ++ns) {
                const float p = __expf(sc[ns][r] - run_max[r]);
                sc[ns][r] = p;
                rs += p;
            }
#pragma unroll
            for (int off = 1; off < 16; off <<= 1)
                rs += __shfl_xor(rs, off, 32);
            run_sum[r] = run_sum[r] * corr[r] + rs;
        }
#pragma unroll
        for (int dsub = 0; dsub < 4; ++dsub)
#pragma unroll
            for (int r = 0; r < 8; ++r)
                o[dsub][r] *= corr[r];

        // ---- P (C layout) -> per-wave LDS -> A layout ----
        {
            unsigned short* P16 = (unsigned short*)&Ps[w][0];
#pragma unroll
            for (int ns = 0; ns < 4; ++ns)
#pragma unroll
                for (int r = 0; r < 8; ++r)
                    P16[(r + hi8) * 64 + ns * 16 + lm] = f2bf(sc[ns][r]);
        }
        asm volatile("s_wait_dscnt 0" ::: "memory");  // same-wave DS in order

        FragAB pf[2];
#pragma unroll
        for (int kc = 0; kc < 2; ++kc)
#pragma unroll
            for (int j = 0; j < 8; ++j)
                pf[kc].u[j] = Ps[w][lm * 32 + kc * 16 + frag_di(j, klo4)];

        // ---- O += P * V : 4 d groups x 2 key-chunks = 8 WMMAs ----
#pragma unroll
        for (int dsub = 0; dsub < 4; ++dsub) {
            const int dc = dsub * 16 + lm;
#pragma unroll
            for (int kc = 0; kc < 2; ++kc) {
                FragAB vf;
#pragma unroll
                for (int j = 0; j < 8; ++j)
                    vf.u[j] = Vsb[dc * 32 + kc * 16 + frag_di(j, klo4)];
                o[dsub] = wmma_bf16(pf[kc], vf, o[dsub]);
            }
        }
        __syncthreads();                               // reads done before buf reuse
    }

    // ---- normalize + store attn out bf16 [B, S, H*64] ----
#pragma unroll
    for (int r = 0; r < 8; ++r) {
        const float inv = 1.0f / run_sum[r];
        const int q = qbase + r + hi8;
        unsigned short* Arow = Ao + ((size_t)b * SEQ + q) * D_MODEL + h * HEAD_DIM;
#pragma unroll
        for (int dsub = 0; dsub < 4; ++dsub)
            Arow[dsub * 16 + lm] = f2bf(o[dsub][r] * inv);
    }
}

// ---------------------------------------------------------------------------
extern "C" void kernel_launch(void* const* d_in, const int* in_sizes, int n_in,
                              void* d_out, int out_size, void* d_ws, size_t ws_size,
                              hipStream_t stream) {
    const float* x    = (const float*)d_in[0];
    const float* Wqkv = (const float*)d_in[1];
    const float* bqkv = (const float*)d_in[2];
    const float* Wout = (const float*)d_in[3];
    const float* bout = (const float*)d_in[4];
    const float* relb = (const float*)d_in[5];
    float* out = (float*)d_out;

    const int M = BATCH * SEQ;                                          // 4096
    const size_t nX   = (size_t)M * D_MODEL;                            // 4M
    const size_t nQKV = (size_t)BATCH * N_HEADS * SEQ * HEAD_DIM;       // 4M

    unsigned short* xb    = (unsigned short*)d_ws;
    unsigned short* WqkvT = xb + nX;
    unsigned short* WoutT = WqkvT + (size_t)3 * 1024 * 1024;
    unsigned short* Qw    = WoutT + (size_t)1024 * 1024;
    unsigned short* Kw    = Qw + nQKV;
    unsigned short* VTw   = Kw + nQKV;
    unsigned short* Aw    = VTw + nQKV;

    // K0: precision conversion + weight transposes
    prep_kernel<<<(int)(nX / 256), 256, 0, stream>>>(x, Wqkv, Wout, xb, WqkvT, WoutT);

    // K1: QKV projection
    {
        const int N = 3 * D_MODEL, K = D_MODEL;
        const int nTn = N / 128;                      // 24
        gemm_bf16_kernel<0><<<(M / 128) * nTn, 256, 0, stream>>>(
            xb, WqkvT, bqkv, Qw, Kw, VTw, nullptr, M, N, K, nTn);
    }
    // K2: attention
    attn_kernel<<<BATCH * N_HEADS * (SEQ / 64), 128, 0, stream>>>(Qw, Kw, VTw, relb, Aw);

    // K3: output projection
    {
        const int N = D_MODEL, K = D_MODEL;
        const int nTn = N / 128;                      // 8
        gemm_bf16_kernel<1><<<(M / 128) * nTn, 256, 0, stream>>>(
            Aw, WoutT, bout, nullptr, nullptr, nullptr, out, M, N, K, nTn);
    }
}
